// TSConv_80101140070458
// MI455X (gfx1250) — matrix operand order repaired and checked by reading, not compiled, not run on hardware
//
#include <hip/hip_runtime.h>
#include <hip/hip_bf16.h>

typedef __bf16 v16bf __attribute__((ext_vector_type(16)));
typedef __bf16 v8bf  __attribute__((ext_vector_type(8)));
typedef float  v8f   __attribute__((ext_vector_type(8)));

#define B_   16
#define C_   64
#define H_   256
#define T_   256
#define OC_  64
// block computes 4 h-rows x 64 t x 64 oc
#define HB   4
#define TB   64
#define LDS_H 6      // 4 + 2 halo
#define LDS_T 66     // 64 + 2 halo
#define CSTR  72     // padded c-stride (ushorts): 144B/t-slot -> conflict-free
                     // b128 reads (lane bank offset n*36 mod 64 tiles all 64
                     // banks in disjoint 4-bank groups) and 16B alignment.

__device__ __forceinline__ unsigned short f2bf(float f) {
    unsigned u = __float_as_uint(f);
    unsigned r = u + 0x7FFFu + ((u >> 16) & 1u);
    return (unsigned short)(r >> 16);
}

// Repack W[OC][C][3][3] (f32, OIHW) into WMMA A-fragments (bf16) in d_ws.
// Fragment f = ((tap*2 + cc)*4 + octile); each lane owns 16 consecutive bf16
// (32B) matching the 16-bit A 16x32 lane layout:
//   lanes 0-15 : M=lane,    K = 0..7 then 16..23
//   lanes 16-31: M=lane-16, K = 8..15 then 24..31
__global__ __launch_bounds__(256) void tsconv_repack_w(
    const float* __restrict__ W, unsigned short* __restrict__ Wp) {
    const int total = 9 * 2 * 4 * 32 * 16; // 36864
    for (int i = threadIdx.x; i < total; i += 256) {
        int j    = i & 15;
        int lane = (i >> 4) & 31;
        int oct  = (i >> 9) & 3;
        int cc   = (i >> 11) & 1;
        int tap  = i >> 12;
        int kh = tap / 3, kw = tap % 3;
        int M    = lane & 15;
        int base = (lane >= 16) ? 8 : 0;
        int K    = (j < 8) ? (j + base) : (j + 8 + base);
        int oc = oct * 16 + M;
        int c  = cc * 32 + K;
        Wp[i] = f2bf(W[((oc * C_ + c) * 3 + kh) * 3 + kw]);
    }
}

__global__ __launch_bounds__(256) void tsconv_wmma_kernel(
    const float* __restrict__ x,
    const unsigned short* __restrict__ Wp,
    const float* __restrict__ bias,
    float* __restrict__ out) {

    __shared__ unsigned short tile[LDS_H * LDS_T * CSTR]; // 57024 B

    const int tid = threadIdx.x;
    const int gid = blockIdx.x;
    const int tb = gid & 3;            // T/64 = 4
    const int hb = (gid >> 2) & 63;    // H/4  = 64
    const int bi = gid >> 8;           // B    = 16
    const int hbase = hb * HB;
    const int tbase = tb * TB;

    // ---- stage shifted input into LDS as bf16, layout [h][t][c], c-contig ----
    // Temporal shift folded into the DESTINATION slot: x[c][h][t] lands at
    // slot = (t - s) - tbase + 1, s = +1 (c<16), -1 (16<=c<32), 0 else.
    // Global reads stay aligned & coalesced (float4 along t); the only extra
    // rule is conv-SAME zero pad: slots whose t' = t - s falls outside the
    // image must hold 0.  Interior covers slots [1-s, 64-s]; halo pass covers
    // the remaining 2 per (h,c) row -> every slot written exactly once.
    for (int i = tid; i < LDS_H * C_ * (TB / 4); i += 256) {
        int q  = i & 15;             // which float4 along t (lane-fast)
        int r  = i >> 4;
        int hh = r % LDS_H;          // half-waves differ in hh -> fewer LDS
        int c  = r / LDS_H;          //   store bank conflicts
        int hg = hbase + hh - 1;
        int s  = (c < 16) ? 1 : ((c < 32) ? -1 : 0);
        int t0 = tbase + q * 4;
        float4 v = make_float4(0.f, 0.f, 0.f, 0.f);
        if ((unsigned)hg < (unsigned)H_)
            v = *reinterpret_cast<const float4*>(
                x + ((bi * C_ + c) * H_ + hg) * T_ + t0);
        float vv[4] = {v.x, v.y, v.z, v.w};
        unsigned short* dst = &tile[(hh * LDS_T + (q * 4 + 1 - s)) * CSTR + c];
#pragma unroll
        for (int j = 0; j < 4; ++j) {
            float f = vv[j];
            if ((unsigned)(t0 + j - s) >= (unsigned)T_) f = 0.f; // conv t-pad
            dst[j * CSTR] = f2bf(f);
        }
    }
    // halo: 2 extra x-elements per (h,c) row
    for (int i = tid; i < LDS_H * C_ * 2; i += 256) {
        int k  = i & 1;
        int r  = i >> 1;
        int hh = r % LDS_H;
        int c  = r / LDS_H;
        int hg = hbase + hh - 1;
        int s  = (c < 16) ? 1 : ((c < 32) ? -1 : 0);
        int tx;
        if (s == 1)       tx = tbase + 64 + k;
        else if (s == -1) tx = tbase - 2 + k;
        else              tx = k ? (tbase + 64) : (tbase - 1);
        int slot = tx - s - tbase + 1;   // in [0,65]
        float f = 0.f;
        if ((unsigned)hg < (unsigned)H_ && (unsigned)tx < (unsigned)T_)
            f = x[((bi * C_ + c) * H_ + hg) * T_ + tx];
        tile[(hh * LDS_T + slot) * CSTR + c] = f2bf(f);
    }
    __syncthreads();

    // ---- per-wave tile assignment: 4 octiles x 2 tpairs ----
    const int wave  = tid >> 5;
    const int lane  = tid & 31;
    const int oct   = wave & 3;        // which 16 output channels
    const int tpair = wave >> 2;       // which pair of 16-t tiles
    const int n     = lane & 15;       // N index (t within tile)
    const int khalf = lane >> 4;

    // C/D layout: VGPR r, lane l -> M = r + (l>=16 ? 8 : 0), N = l&15.
    // Init accumulators with bias (folds the + b epilogue into C).
    v8f acc[HB * 2];
    float bv[8];
#pragma unroll
    for (int r = 0; r < 8; ++r)
        bv[r] = bias[oct * 16 + khalf * 8 + r];
#pragma unroll
    for (int a = 0; a < HB * 2; ++a)
#pragma unroll
        for (int r = 0; r < 8; ++r)
            acc[a][r] = bv[r];

    // ---- reduction: 9 taps x 2 chunks of 32 channels, A reused over 8 tiles ----
#pragma unroll
    for (int tap = 0; tap < 9; ++tap) {
        const int kh = tap / 3, kw = tap % 3;
#pragma unroll
        for (int cc = 0; cc < 2; ++cc) {
            // A fragment: 32 contiguous bytes per lane, L2/WGP$-resident.
            const v16bf af = *reinterpret_cast<const v16bf*>(
                Wp + ((((tap * 2 + cc) * 4 + oct) * 32 + lane) << 4));
#pragma unroll
            for (int hr = 0; hr < HB; ++hr) {
#pragma unroll
                for (int tt = 0; tt < 2; ++tt) {
                    // B fragment: lanes 0-15 K=c 0..7,16..23, lanes 16-31 K=c 8..15,24..31,
                    // N = lane&15 -> t.  Two explicit 16B halves (odd t-slots
                    // are 16B- but not 32B-aligned with the 144B stride).
                    const int toff = tpair * 32 + tt * 16 + n + kw; // 0..65
                    const unsigned short* bp =
                        tile + ((hr + kh) * LDS_T + toff) * CSTR +
                        cc * 32 + khalf * 8;
                    union { v16bf v; v8bf h[2]; } bf;
                    bf.h[0] = *reinterpret_cast<const v8bf*>(bp);
                    bf.h[1] = *reinterpret_cast<const v8bf*>(bp + 16);
                    acc[hr * 2 + tt] = __builtin_amdgcn_wmma_f32_16x16x32_bf16(
                        false, af, false, bf.v, (short)0, acc[hr * 2 + tt],
                        false, false);
                }
            }
        }
    }

    // ---- store D (bias already accumulated); lanes 0-15 / 16-31 each form a
    // contiguous 64B run per r, per tile ----
#pragma unroll
    for (int hr = 0; hr < HB; ++hr) {
#pragma unroll
        for (int tt = 0; tt < 2; ++tt) {
            const int tg = tbase + tpair * 32 + tt * 16 + n;
            const int hg = hbase + hr;
#pragma unroll
            for (int r = 0; r < 8; ++r) {
                const int oc = oct * 16 + khalf * 8 + r;
                out[((bi * OC_ + oc) * H_ + hg) * T_ + tg] = acc[hr * 2 + tt][r];
            }
        }
    }
}

extern "C" void kernel_launch(void* const* d_in, const int* in_sizes, int n_in,
                              void* d_out, int out_size, void* d_ws, size_t ws_size,
                              hipStream_t stream) {
    const float* x = (const float*)d_in[0];   // [16,64,256,256] f32
    const float* W = (const float*)d_in[1];   // [64,64,3,3] f32
    const float* b = (const float*)d_in[2];   // [64] f32
    float* out = (float*)d_out;
    unsigned short* Wp = (unsigned short*)d_ws; // 36864 * 2 bytes

    tsconv_repack_w<<<1, 256, 0, stream>>>(W, Wp);

    const int grid = B_ * (H_ / HB) * (T_ / TB); // 16*64*4 = 4096
    tsconv_wmma_kernel<<<grid, 256, 0, stream>>>(x, Wp, b, out);
}
